// MaskRCNN_22359599743290
// MI455X (gfx1250) — matrix operand order repaired and checked by reading, not compiled, not run on hardware
//
#include <hip/hip_runtime.h>
#include <hip/hip_bf16.h>
#include <math.h>

typedef __attribute__((ext_vector_type(16))) _Float16 v16h;
typedef __attribute__((ext_vector_type(8)))  _Float16 v8h;
typedef __attribute__((ext_vector_type(8)))  float    v8f;
typedef __attribute__((ext_vector_type(4)))  float    v4f;

#define IMG   800
#define FH    50
#define N_PRE 12000
#define NSORT 32768
#define NWORD 375          // 12000/32
#define N_ROI 128
#define POOL  7

// ---------------------------------------------------------------------------
// Generic WMMA GEMM: C[M][N] = relu?(A[M][K](f16) * B + bias)
//   B is f32 in HBM: bT==0 -> B layout [N][K] (conv/fc weights, O-major)
//                    bT==1 -> B layout [K][N] (deconv natural layout)
// Requirements (all call sites satisfy): K % 64 == 0, N % 64 == 0.
// Tile: 64x64, BK=64, 128 threads (4 waves); wave w owns rows [16w,16w+16).
// f32 weights are converted to f16 while staging into LDS (single HBM pass).
// LDS keeps B as [n][k] so both A and B fragments are contiguous b128 loads.
// ---------------------------------------------------------------------------
#define BM 64
#define BN 64
#define BK 64
#define LDP 72   // row pitch in halves (144B, 16B-aligned)

__device__ inline v16h ld_frag(const _Float16* p0, const _Float16* p1)
{
    v8h a = *(const v8h*)p0;
    v8h b = *(const v8h*)p1;
    return __builtin_shufflevector(a, b, 0,1,2,3,4,5,6,7,8,9,10,11,12,13,14,15);
}

__launch_bounds__(128)
__global__ void wmma_gemm_k(const _Float16* __restrict__ A,
                            const float* __restrict__ Bw,
                            int M, int N, int K, int bT,
                            const float* __restrict__ bias, int relu,
                            float* __restrict__ C, _Float16* __restrict__ C16)
{
    __shared__ __align__(16) _Float16 As[BM][LDP];
    __shared__ __align__(16) _Float16 Bs[BN][LDP];   // [n][k]

    const int m0   = blockIdx.x * BM;
    const int n0   = blockIdx.y * BN;
    const int tid  = threadIdx.x;
    const int wave = tid >> 5;
    const int lane = tid & 31;

    v8f acc0 = {0.f,0.f,0.f,0.f,0.f,0.f,0.f,0.f};
    v8f acc1 = acc0, acc2 = acc0, acc3 = acc0;

    const int arow = wave * 16 + (lane & 15);
    const int koff = (lane < 16) ? 0 : 8;
    const int col  = lane & 15;
    const int kb   = (lane < 16) ? 0 : 16;

    for (int k0 = 0; k0 < K; k0 += BK) {
        // prefetch next k-tile (emits global_prefetch_b8)
        int knext = k0 + BK;
        if (knext < K) {
            int r  = tid >> 3, cv = tid & 7;
            if (m0 + r < M)
                __builtin_prefetch(A + (size_t)(m0 + r) * K + knext + cv * 8, 0, 0);
            if (!bT)
                __builtin_prefetch(Bw + (size_t)(n0 + r) * K + knext + cv * 16, 0, 0);
        }

        // ---- stage A tile: 64x64 halves = 512 v8h vectors, 4 per thread ----
        #pragma unroll
        for (int it = 0; it < 4; ++it) {
            int v  = it * 128 + tid;
            int r  = v >> 3, cv = v & 7;          // 8 vectors per row
            int mm = m0 + r;
            v8h val = {0,0,0,0,0,0,0,0};
            if (mm < M) val = *(const v8h*)(A + (size_t)mm * K + k0 + cv * 8);
            *(v8h*)&As[r][cv * 8] = val;
        }
        // ---- stage B tile into [n][k], f32 -> f16 in-flight ----
        if (!bT) {
            // weights [N][K]: contiguous along k -> vector loads
            #pragma unroll
            for (int it = 0; it < 4; ++it) {
                int v  = it * 128 + tid;
                int nn = v >> 3, cv = v & 7;
                const float* src = Bw + (size_t)(n0 + nn) * K + k0 + cv * 8;
                v4f f0 = *(const v4f*)src;
                v4f f1 = *(const v4f*)(src + 4);
                v8h h;
                #pragma unroll
                for (int q = 0; q < 4; ++q) {
                    h[q]     = (_Float16)f0[q];
                    h[4 + q] = (_Float16)f1[q];
                }
                *(v8h*)&Bs[nn][cv * 8] = h;
            }
        } else {
            // [K][N] layout (deconv only, K=512): strided gather
            #pragma unroll
            for (int it = 0; it < 32; ++it) {
                int e  = it * 128 + tid;
                int nn = e >> 6, kk = e & 63;
                Bs[nn][kk] = (_Float16)Bw[(size_t)(k0 + kk) * N + n0 + nn];
            }
        }
        __syncthreads();

        // ---- 8 WMMAs per tile: 2 k-steps x 4 n-subtiles ----
        #pragma unroll
        for (int ks = 0; ks < BK; ks += 32) {
            // A frag (16x32): lanes 0-15 K {0-7,16-23}; lanes 16-31 K {8-15,24-31}
            v16h af = ld_frag(&As[arow][ks + koff], &As[arow][ks + 16 + koff]);
            v16h bf;
            bf = ld_frag(&Bs[0 * 16 + col][ks + koff], &Bs[0 * 16 + col][ks + 16 + koff]);
            acc0 = __builtin_amdgcn_wmma_f32_16x16x32_f16(false, af, false, bf, (short)0, acc0, false, false);
            bf = ld_frag(&Bs[1 * 16 + col][ks + koff], &Bs[1 * 16 + col][ks + 16 + koff]);
            acc1 = __builtin_amdgcn_wmma_f32_16x16x32_f16(false, af, false, bf, (short)0, acc1, false, false);
            bf = ld_frag(&Bs[2 * 16 + col][ks + koff], &Bs[2 * 16 + col][ks + 16 + koff]);
            acc2 = __builtin_amdgcn_wmma_f32_16x16x32_f16(false, af, false, bf, (short)0, acc2, false, false);
            bf = ld_frag(&Bs[3 * 16 + col][ks + koff], &Bs[3 * 16 + col][ks + 16 + koff]);
            acc3 = __builtin_amdgcn_wmma_f32_16x16x32_f16(false, af, false, bf, (short)0, acc3, false, false);
        }
        __syncthreads();
    }

    // epilogue per ISA 32-bit C/D layout: VGPR r -> M = r (+8 for lanes 16-31)
    const int rbase = m0 + wave * 16 + ((lane >> 4) << 3);
    v8f accs[4] = {acc0, acc1, acc2, acc3};
    #pragma unroll
    for (int t = 0; t < 4; ++t) {
        int n = n0 + t * 16 + col;
        float bv = bias ? bias[n] : 0.f;
        #pragma unroll
        for (int r = 0; r < 8; ++r) {
            int m = rbase + r;
            if (m >= M) continue;
            float v = accs[t][r] + bv;
            if (relu && v < 0.f) v = 0.f;
            if (C)   C[(size_t)m * N + n]   = v;
            if (C16) C16[(size_t)m * N + n] = (_Float16)v;
        }
    }
}

// ---------------------------------------------------------------------------
// NCHW f32 (512,50,50) -> [spatial(2500)][C(512)] f16
// ---------------------------------------------------------------------------
__global__ void to_mc_k(const float* __restrict__ feat, _Float16* __restrict__ dst)
{
    int t = blockIdx.x * blockDim.x + threadIdx.x;
    if (t >= FH * FH * 512) return;
    int c = t & 511, m = t >> 9;
    dst[t] = (_Float16)feat[c * (FH * FH) + m];
}

// ---------------------------------------------------------------------------
// im2col 3x3 pad1: src [B*H*W][C] f16 -> A [B*H*W][C*9] f16 (k = c*9+ky*3+kx)
// ---------------------------------------------------------------------------
__global__ void im2col_k(const _Float16* __restrict__ src, _Float16* __restrict__ A,
                         int B, int H, int W, int C)
{
    int t = blockIdx.x * blockDim.x + threadIdx.x;
    if (t >= B * H * W * C) return;
    int c = t % C, m = t / C;
    int x = m % W, y = (m / W) % H;
    int bbase = (m / (W * H)) * (W * H);
    size_t arow = (size_t)m * (C * 9) + (size_t)c * 9;
    #pragma unroll
    for (int ky = 0; ky < 3; ++ky) {
        int yy = y + ky - 1;
        #pragma unroll
        for (int kx = 0; kx < 3; ++kx) {
            int xx = x + kx - 1;
            _Float16 v = (_Float16)0.f;
            if (yy >= 0 && yy < H && xx >= 0 && xx < W)
                v = src[((size_t)bbase + yy * W + xx) * C + c];
            A[arow + ky * 3 + kx] = v;
        }
    }
}

// ---------------------------------------------------------------------------
// RPN 1x1 heads: scores (2500x18) + offsets (2500x36) from h16 [2500][512]
// ---------------------------------------------------------------------------
__global__ void rpn_heads_k(const _Float16* __restrict__ h16,
                            const float* __restrict__ cls_w, const float* __restrict__ cls_b,
                            const float* __restrict__ reg_w, const float* __restrict__ reg_b,
                            float* __restrict__ scores, float* __restrict__ offs)
{
    int t = blockIdx.x * blockDim.x + threadIdx.x;
    if (t >= FH * FH * 54) return;
    int m = t / 54, o = t % 54;
    const _Float16* hp = h16 + (size_t)m * 512;
    const float* w;
    float acc;
    if (o < 18) { w = cls_w + o * 512;        acc = cls_b[o]; }
    else        { w = reg_w + (o - 18) * 512; acc = reg_b[o - 18]; }
    for (int k = 0; k < 512; ++k) acc += (float)hp[k] * w[k];
    if (o < 18) scores[m * 18 + o]      = acc;
    else        offs[m * 36 + (o - 18)] = acc;
}

// ---------------------------------------------------------------------------
// Anchor generation + box decode + sort-key init (padded to 32768)
// ---------------------------------------------------------------------------
__global__ void decode_k(const float* __restrict__ scores, const float* __restrict__ offs,
                         float* __restrict__ bbox, float* __restrict__ keys, int* __restrict__ idx)
{
    int i = blockIdx.x * blockDim.x + threadIdx.x;
    if (i >= NSORT) return;
    idx[i] = i;
    if (i >= FH * FH * 9) { keys[i] = -3.4e38f; return; }
    int t  = i % 9, m = i / 9;
    int ix = m % FH, iy = m / FH;
    const float rv[3] = {0.5f, 1.f, 2.f};
    const float av[3] = {80.f, 400.f, 1000.f};
    float r = rv[t / 3], a = av[t % 3];
    float hh = sqrtf(a / r) * 0.5f;
    float ww = sqrtf(a * r) * 0.5f;
    float cy = 8.f + 16.f * iy, cx = 8.f + 16.f * ix;
    float h = 2.f * hh, w = 2.f * ww;
    const float* o = offs + m * 36 + t * 4;
    float dy = o[0], dx = o[1];
    float dh = 0.5f * __expf(o[2]);
    float dw = 0.5f * __expf(o[3]);
    bbox[i * 4 + 0] = fminf(fmaxf(cy + h * (dy - dh), 0.f), (float)IMG);
    bbox[i * 4 + 1] = fminf(fmaxf(cx + w * (dx - dw), 0.f), (float)IMG);
    bbox[i * 4 + 2] = fminf(fmaxf(cy + h * (dy + dh), 0.f), (float)IMG);
    bbox[i * 4 + 3] = fminf(fmaxf(cx + w * (dx + dw), 0.f), (float)IMG);
    keys[i] = scores[m * 18 + t * 2 + 1];
}

// bitonic compare-exchange (overall descending)
__global__ void bitonic_k(float* __restrict__ key, int* __restrict__ idx, int j, int k)
{
    int i = blockIdx.x * blockDim.x + threadIdx.x;
    int ixj = i ^ j;
    if (i < NSORT && ixj > i) {
        bool asc = (i & k) != 0;
        float ki = key[i], kj = key[ixj];
        bool swap = asc ? (ki > kj) : (ki < kj);
        if (swap) {
            key[i] = kj; key[ixj] = ki;
            int tt = idx[i]; idx[i] = idx[ixj]; idx[ixj] = tt;
        }
    }
}

__global__ void gather_sorted_k(const float* __restrict__ bbox, const int* __restrict__ idx,
                                float* __restrict__ boxes)
{
    int i = blockIdx.x * blockDim.x + threadIdx.x;
    if (i >= N_PRE) return;
    int s = idx[i];
    boxes[i * 4 + 0] = bbox[s * 4 + 0];
    boxes[i * 4 + 1] = bbox[s * 4 + 1];
    boxes[i * 4 + 2] = bbox[s * 4 + 2];
    boxes[i * 4 + 3] = bbox[s * 4 + 3];
}

// IoU suppression bitmask: mask[i][w] bit b -> iou(box_i, box_{32w+b}) >= 0.7
__global__ void nms_mask_k(const float* __restrict__ boxes, unsigned* __restrict__ mask)
{
    int t = blockIdx.x * blockDim.x + threadIdx.x;
    if (t >= N_PRE * NWORD) return;
    int i = t / NWORD, w = t % NWORD;
    float y1 = boxes[i*4], x1 = boxes[i*4+1], y2 = boxes[i*4+2], x2 = boxes[i*4+3];
    float ai = (y2 - y1) * (x2 - x1);
    unsigned bits = 0;
    for (int b = 0; b < 32; ++b) {
        int j = w * 32 + b;
        float jy1 = boxes[j*4], jx1 = boxes[j*4+1], jy2 = boxes[j*4+2], jx2 = boxes[j*4+3];
        float ty = fmaxf(y1, jy1), tx = fmaxf(x1, jx1);
        float by = fminf(y2, jy2), bx = fminf(x2, jx2);
        float cap = (ty < by && tx < bx) ? (by - ty) * (bx - tx) : 0.f;
        float aj  = (jy2 - jy1) * (jx2 - jx1);
        float iou = cap / (ai + aj - cap);
        if (iou >= 0.7f) bits |= (1u << b);
    }
    mask[(size_t)i * NWORD + w] = bits;
}

// sequential NMS scan (single block), emits 128 ROI boxes (padded w/ index 0)
__global__ void nms_scan_k(const unsigned* __restrict__ mask,
                           const float* __restrict__ boxes, float* __restrict__ rois)
{
    __shared__ unsigned removed[NWORD];
    __shared__ int list[N_ROI];
    __shared__ int cnt;
    __shared__ int curKeep;
    int tid = threadIdx.x;
    if (tid < NWORD) removed[tid] = 0u;
    if (tid == 0) cnt = 0;
    __syncthreads();
    for (int i = 0; i < N_PRE; ++i) {
        if (tid == 0) {
            int kept = !((removed[i >> 5] >> (i & 31)) & 1u);
            curKeep = kept;
            if (kept) { if (cnt < N_ROI) list[cnt] = i; cnt++; }
        }
        __syncthreads();
        if (curKeep && tid < NWORD) removed[tid] |= mask[(size_t)i * NWORD + tid];
        __syncthreads();
    }
    if (tid < N_ROI) {
        int j = (tid < cnt && tid < N_ROI) ? list[tid] : 0;
        rois[tid * 4 + 0] = boxes[j * 4 + 0];
        rois[tid * 4 + 1] = boxes[j * 4 + 1];
        rois[tid * 4 + 2] = boxes[j * 4 + 2];
        rois[tid * 4 + 3] = boxes[j * 4 + 3];
    }
}

// ---------------------------------------------------------------------------
// ROI pooling (bilinear, 7x7) -> pool_ms [(r*49+p)][512] f16 (mask branch)
//                               A6 [r][c*49+p] f16 (fc6 input, C-major flatten)
// ---------------------------------------------------------------------------
__device__ inline float roi_tap(const float* feat, int c, int y, int x)
{
    bool valid = (x >= 0) && (x < FH) && (y >= 0) && (y < FH);
    int yc = min(max(y, 0), FH - 1), xc = min(max(x, 0), FH - 1);
    float v = feat[c * (FH * FH) + yc * FH + xc];
    return valid ? v : 0.f;
}

__global__ void roi_pool_k(const float* __restrict__ feat, const float* __restrict__ rois,
                           _Float16* __restrict__ pool_ms, _Float16* __restrict__ A6)
{
    int t = blockIdx.x * blockDim.x + threadIdx.x;
    if (t >= N_ROI * 512 * 49) return;
    int p = t % 49, c = (t / 49) % 512, r = t / (49 * 512);
    int iy = p / 7, ix = p % 7;
    const float* roi = rois + r * 4;
    float sc = (float)FH / (float)IMG;
    float top = roi[0] * sc, left = roi[1] * sc, bottom = roi[2] * sc, right = roi[3] * sc;
    float t0a = (bottom - top) / 49.f, t0c = (top + bottom - 49.f) / 49.f;
    float t1b = (right - left) / 49.f, t1c = (left + right - 49.f) / 49.f;
    float bx = -1.f + 2.f * ix / 6.f;
    float by = -1.f + 2.f * iy / 6.f;
    float pxf = (t0a * bx + t0c + 1.f) * 0.5f * 49.f;
    float pyf = (t1b * by + t1c + 1.f) * 0.5f * 49.f;
    float x0f = floorf(pxf), y0f = floorf(pyf);
    float wx1 = pxf - x0f, wx0 = 1.f - wx1;
    float wy1 = pyf - y0f, wy0 = 1.f - wy1;
    int x0 = (int)x0f, y0 = (int)y0f;
    float v = roi_tap(feat, c, y0,     x0    ) * wy0 * wx0
            + roi_tap(feat, c, y0,     x0 + 1) * wy0 * wx1
            + roi_tap(feat, c, y0 + 1, x0    ) * wy1 * wx0
            + roi_tap(feat, c, y0 + 1, x0 + 1) * wy1 * wx1;
    pool_ms[((size_t)r * 49 + p) * 512 + c]         = (_Float16)v;
    A6[(size_t)r * (512 * 49) + (size_t)c * 49 + p] = (_Float16)v;
}

// fc head outputs: roi_cls_locs (128x8) -> out[0..1023], roi_scores (128x2) -> out[1024..]
__global__ void fc_heads_k(const _Float16* __restrict__ f7,
                           const float* __restrict__ wl, const float* __restrict__ bl,
                           const float* __restrict__ wsc, const float* __restrict__ bsc,
                           float* __restrict__ out)
{
    int t = blockIdx.x * blockDim.x + threadIdx.x;
    if (t >= N_ROI * 10) return;
    int r = t / 10, o = t % 10;
    const float* w; float b; int dst;
    if (o < 8) { w = wl + o * 4096;        b = bl[o];      dst = r * 8 + o; }
    else       { w = wsc + (o - 8) * 4096; b = bsc[o - 8]; dst = 1024 + r * 2 + (o - 8); }
    float s = b;
    const _Float16* fp = f7 + (size_t)r * 4096;
    for (int k = 0; k < 4096; ++k) s += (float)fp[k] * w[k];
    out[dst] = s;
}

// deconv epilogue: Y [6272][1024] f32 -> m2_ms [(r*196 + (2h+a)*14 + 2w+b)][256] f16
__global__ void deconv_scatter_k(const float* __restrict__ Y, const float* __restrict__ db,
                                 _Float16* __restrict__ m2)
{
    int t = blockIdx.x * blockDim.x + threadIdx.x;
    if (t >= N_ROI * 49 * 1024) return;
    int q = t % 1024, m = t / 1024;
    int o = q >> 2, a = (q >> 1) & 1, b = q & 1;
    int r = m / 49, p = m % 49, h = p / 7, w = p % 7;
    float v = Y[t] + db[o];
    if (v < 0.f) v = 0.f;
    size_t dst = ((size_t)r * 196 + (2 * h + a) * 14 + (2 * w + b)) * 256 + o;
    m2[dst] = (_Float16)v;
}

// final 3x3 conv 256 -> 2 on 14x14, writes roi_mask to d_out (pre-offset ptr)
__global__ void c22_k(const _Float16* __restrict__ m21, const float* __restrict__ w,
                      const float* __restrict__ bb, float* __restrict__ out)
{
    int t = blockIdx.x * blockDim.x + threadIdx.x;
    if (t >= N_ROI * 2 * 196) return;
    int p = t % 196, oc = (t / 196) % 2, r = t / 392;
    int y = p / 14, x = p % 14;
    float acc = bb[oc];
    for (int c = 0; c < 256; ++c) {
        #pragma unroll
        for (int ky = 0; ky < 3; ++ky) {
            int yy = y + ky - 1;
            if (yy < 0 || yy >= 14) continue;
            #pragma unroll
            for (int kx = 0; kx < 3; ++kx) {
                int xx = x + kx - 1;
                if (xx < 0 || xx >= 14) continue;
                acc += (float)m21[((size_t)r * 196 + yy * 14 + xx) * 256 + c]
                     * w[(oc * 256 + c) * 9 + ky * 3 + kx];
            }
        }
    }
    out[(size_t)(r * 2 + oc) * 196 + p] = acc;
}

// ---------------------------------------------------------------------------
extern "C" void kernel_launch(void* const* d_in, const int* in_sizes, int n_in,
                              void* d_out, int out_size, void* d_ws, size_t ws_size,
                              hipStream_t stream)
{
    (void)in_sizes; (void)n_in; (void)out_size; (void)ws_size;
    const float* features  = (const float*)d_in[0];
    const float* inter_w   = (const float*)d_in[1];
    const float* inter_b   = (const float*)d_in[2];
    const float* cls_w     = (const float*)d_in[3];
    const float* cls_b     = (const float*)d_in[4];
    const float* reg_w     = (const float*)d_in[5];
    const float* reg_b     = (const float*)d_in[6];
    const float* fc6_w     = (const float*)d_in[7];
    const float* fc6_b     = (const float*)d_in[8];
    const float* fc7_w     = (const float*)d_in[9];
    const float* fc7_b     = (const float*)d_in[10];
    const float* cls_loc_w = (const float*)d_in[11];
    const float* cls_loc_b = (const float*)d_in[12];
    const float* score_w   = (const float*)d_in[13];
    const float* score_b   = (const float*)d_in[14];
    const float* c11_w     = (const float*)d_in[15];
    const float* c11_b     = (const float*)d_in[16];
    const float* c12_w     = (const float*)d_in[17];
    const float* c12_b     = (const float*)d_in[18];
    const float* deconv_w  = (const float*)d_in[19];
    const float* deconv_b  = (const float*)d_in[20];
    const float* c21_w     = (const float*)d_in[21];
    const float* c21_b     = (const float*)d_in[22];
    const float* c22_w     = (const float*)d_in[23];
    const float* c22_b     = (const float*)d_in[24];
    float* out = (float*)d_out;

    // ---- scratch carve (bump allocator, 256B aligned) ----
    char* base = (char*)d_ws;
    size_t off = 0;
    auto alloc = [&](size_t bytes) -> void* {
        void* r = base + off;
        off = (off + bytes + 255) & ~(size_t)255;
        return r;
    };
    _Float16* feat_ms = (_Float16*)alloc((size_t)2500 * 512 * 2);
    _Float16* h16     = (_Float16*)alloc((size_t)2500 * 512 * 2);
    float*    scoresB = (float*)   alloc((size_t)2500 * 18 * 4);
    float*    offsB   = (float*)   alloc((size_t)2500 * 36 * 4);
    float*    bbox    = (float*)   alloc((size_t)22500 * 4 * 4);
    float*    keys    = (float*)   alloc((size_t)NSORT * 4);
    int*      sidx    = (int*)     alloc((size_t)NSORT * 4);
    float*    boxesS  = (float*)   alloc((size_t)N_PRE * 4 * 4);
    unsigned* nmsmask = (unsigned*)alloc((size_t)N_PRE * NWORD * 4);
    float*    rois    = (float*)   alloc((size_t)N_ROI * 4 * 4);
    _Float16* pool_ms = (_Float16*)alloc((size_t)6272 * 512 * 2);
    _Float16* A6      = (_Float16*)alloc((size_t)N_ROI * 25088 * 2);
    _Float16* f6      = (_Float16*)alloc((size_t)N_ROI * 4096 * 2);
    _Float16* f7      = (_Float16*)alloc((size_t)N_ROI * 4096 * 2);
    _Float16* m11     = (_Float16*)alloc((size_t)6272 * 512 * 2);
    _Float16* m12     = (_Float16*)alloc((size_t)6272 * 512 * 2);
    float*    Ydec    = (float*)   alloc((size_t)6272 * 1024 * 4);
    _Float16* m2_ms   = (_Float16*)alloc((size_t)25088 * 256 * 2);
    _Float16* m21     = (_Float16*)alloc((size_t)25088 * 256 * 2);
    _Float16* A_rpn   = (_Float16*)alloc((size_t)2500 * 4608 * 2);
    _Float16* A_mask  = (_Float16*)alloc((size_t)25088 * 2304 * 2); // >= 6272*4608

    auto blks = [](long n) { return (unsigned)((n + 255) / 256); };

    // ---- RPN conv (implicit GEMM, WMMA) ----
    to_mc_k<<<blks(2500L * 512), 256, 0, stream>>>(features, feat_ms);
    im2col_k<<<blks(2500L * 512), 256, 0, stream>>>(feat_ms, A_rpn, 1, FH, FH, 512);
    wmma_gemm_k<<<dim3((2500 + BM - 1) / BM, 512 / BN), 128, 0, stream>>>(
        A_rpn, inter_w, 2500, 512, 4608, 0, inter_b, 1, nullptr, h16);
    rpn_heads_k<<<blks(2500L * 54), 256, 0, stream>>>(
        h16, cls_w, cls_b, reg_w, reg_b, scoresB, offsB);

    // ---- decode + top-k (bitonic desc over 32768) ----
    decode_k<<<blks(NSORT), 256, 0, stream>>>(scoresB, offsB, bbox, keys, sidx);
    for (int k = 2; k <= NSORT; k <<= 1)
        for (int j = k >> 1; j > 0; j >>= 1)
            bitonic_k<<<blks(NSORT), 256, 0, stream>>>(keys, sidx, j, k);
    gather_sorted_k<<<blks(N_PRE), 256, 0, stream>>>(bbox, sidx, boxesS);

    // ---- NMS ----
    nms_mask_k<<<blks((long)N_PRE * NWORD), 256, 0, stream>>>(boxesS, nmsmask);
    nms_scan_k<<<1, 384, 0, stream>>>(nmsmask, boxesS, rois);

    // ---- ROI pool ----
    roi_pool_k<<<blks((long)N_ROI * 512 * 49), 256, 0, stream>>>(features, rois, pool_ms, A6);

    // ---- FC head (WMMA; fc6 reads 411MB f32 weights once, converts in-kernel) ----
    wmma_gemm_k<<<dim3(N_ROI / BM, 4096 / BN), 128, 0, stream>>>(
        A6, fc6_w, N_ROI, 4096, 25088, 0, fc6_b, 1, nullptr, f6);
    wmma_gemm_k<<<dim3(N_ROI / BM, 4096 / BN), 128, 0, stream>>>(
        f6, fc7_w, N_ROI, 4096, 4096, 0, fc7_b, 1, nullptr, f7);
    fc_heads_k<<<blks((long)N_ROI * 10), 256, 0, stream>>>(
        f7, cls_loc_w, cls_loc_b, score_w, score_b, out);

    // ---- mask branch ----
    im2col_k<<<blks(6272L * 512), 256, 0, stream>>>(pool_ms, A_mask, N_ROI, 7, 7, 512);
    wmma_gemm_k<<<dim3((6272 + BM - 1) / BM, 512 / BN), 128, 0, stream>>>(
        A_mask, c11_w, 6272, 512, 4608, 0, c11_b, 1, nullptr, m11);
    im2col_k<<<blks(6272L * 512), 256, 0, stream>>>(m11, A_mask, N_ROI, 7, 7, 512);
    wmma_gemm_k<<<dim3((6272 + BM - 1) / BM, 512 / BN), 128, 0, stream>>>(
        A_mask, c12_w, 6272, 512, 4608, 0, c12_b, 1, nullptr, m12);
    // deconv as GEMM: B layout [K=512][N=1024] (natural deconv_w flat), bT=1
    wmma_gemm_k<<<dim3((6272 + BM - 1) / BM, 1024 / BN), 128, 0, stream>>>(
        m12, deconv_w, 6272, 1024, 512, 1, nullptr, 0, Ydec, nullptr);
    deconv_scatter_k<<<blks(6272L * 1024), 256, 0, stream>>>(Ydec, deconv_b, m2_ms);
    im2col_k<<<blks(25088L * 256), 256, 0, stream>>>(m2_ms, A_mask, N_ROI, 14, 14, 256);
    wmma_gemm_k<<<dim3((25088 + BM - 1) / BM, 256 / BN), 128, 0, stream>>>(
        A_mask, c21_w, 25088, 256, 2304, 0, c21_b, 1, nullptr, m21);
    c22_k<<<blks((long)N_ROI * 2 * 196), 256, 0, stream>>>(m21, c22_w, c22_b, out + 1280);
}
